// Llama4VisionAttention_56985626083813
// MI455X (gfx1250) — matrix operand order repaired and checked by reading, not compiled, not run on hardware
//
#include <hip/hip_runtime.h>

// ---------------------------------------------------------------------------
// Llama4 vision attention on gfx1250 (MI455X), bf16 WMMA pipeline.
//   hidden[32,577,1408] -> QKV GEMM(+bias,+RoPE) -> flash attention -> O-proj
// All GEMMs use v_wmma_f32_16x16x32_bf16 (wave32, 16x16 tiles, K=32 steps).
// ---------------------------------------------------------------------------

#define HIDDEN  1408
#define HEADS   16
#define HD      88
#define SEQ     577
#define BATCH   32
#define MTOT    (BATCH*SEQ)      // 18464 rows
#define NQKV    (3*HIDDEN)       // 4224
#define SP      608              // seq padded to multiple of 32
#define DP      96               // head dim padded to multiple of 32
#define QTILES  37               // ceil(577/16)
#define SCALE_F 0.1066003581778052f   // 88^-0.5

typedef __attribute__((ext_vector_type(16))) __bf16 v16bf;
typedef __attribute__((ext_vector_type(8)))  float  v8f;

union F16Frag { v16bf v; uint4 q[2]; };

__device__ __forceinline__ unsigned short f2bf(float f) {
    unsigned u = __float_as_uint(f);
    u += 0x7fffu + ((u >> 16) & 1u);          // round-to-nearest-even
    return (unsigned short)(u >> 16);
}

__device__ __forceinline__ v8f bf16_wmma(v16bf a, v16bf b, v8f c) {
    return __builtin_amdgcn_wmma_f32_16x16x32_bf16(false, a, false, b,
                                                   (short)0, c, false, false);
}

// A operand (16x32 bf16) from row-major [M][K] storage.
// Lane L: row = M-base + (L&15); holds K[16*half .. +16) (contiguous)
__device__ __forceinline__ v16bf wmma_ldA(const unsigned short* __restrict__ A, int ldk,
                                          int row0, int k, int lane, int rowMax) {
    int half = lane >> 4;
    int r = row0 + (lane & 15);
    if (r >= rowMax) r = rowMax - 1;           // clamp (pad rows unused downstream)
    const unsigned short* p = A + (size_t)r * ldk + k + 16 * half;
    F16Frag f;
    f.q[0] = *(const uint4*)p;
    f.q[1] = *(const uint4*)(p + 8);
    return f.v;
}

// B operand (32x16 bf16) from row-major "transposed" [N][K] storage.
// Lane L: col = N-base + (L&15); holds K[16*half .. +16) (32 contiguous bytes).
__device__ __forceinline__ v16bf wmma_ldBT(const unsigned short* __restrict__ BT, int ldk,
                                           int col0, int k, int lane) {
    int half = lane >> 4;
    const unsigned short* p = BT + (size_t)(col0 + (lane & 15)) * ldk + k + 16 * half;
    F16Frag f;
    f.q[0] = *(const uint4*)p;
    f.q[1] = *(const uint4*)(p + 8);
    return f.v;
}

// ---------------------------------------------------------------------------
// Prep kernels
// ---------------------------------------------------------------------------
__global__ void zero_kernel(uint4* __restrict__ p, size_t n16) {
    size_t i = (size_t)blockIdx.x * blockDim.x + threadIdx.x;
    size_t stride = (size_t)gridDim.x * blockDim.x;
    uint4 z; z.x = 0; z.y = 0; z.z = 0; z.w = 0;
    for (; i < n16; i += stride) p[i] = z;
}

__global__ void cvt_hidden_kernel(const float* __restrict__ src,
                                  unsigned short* __restrict__ dst, int n4) {
    int i = blockIdx.x * blockDim.x + threadIdx.x;
    if (i < n4) {
        float4 f = ((const float4*)src)[i];
        ushort4 o;
        o.x = f2bf(f.x); o.y = f2bf(f.y); o.z = f2bf(f.z); o.w = f2bf(f.w);
        ((ushort4*)dst)[i] = o;
    }
}

// W[K][N] f32 -> WT[N][K] bf16 via 32x32 LDS tile (coalesced both sides)
__global__ __launch_bounds__(256) void transpose_cvt_kernel(
        const float* __restrict__ W, unsigned short* __restrict__ WT, int K, int N) {
    __shared__ float t[32][33];
    int n0 = blockIdx.x * 32, k0 = blockIdx.y * 32;
    int tx = threadIdx.x, ty = threadIdx.y;   // 32 x 8
#pragma unroll
    for (int i = 0; i < 4; i++)
        t[ty + 8 * i][tx] = W[(size_t)(k0 + ty + 8 * i) * N + n0 + tx];
    __syncthreads();
#pragma unroll
    for (int i = 0; i < 4; i++)
        WT[(size_t)(n0 + ty + 8 * i) * K + k0 + tx] = f2bf(t[tx][ty + 8 * i]);
}

// RoPE tables [SP][44]: 2D (24x24) position rope; row 576 = class token (idx<0 -> 0)
__global__ void rope_table_kernel(float* __restrict__ cosT, float* __restrict__ sinT) {
    int s = blockIdx.x;
    int p = threadIdx.x;
    if (p >= 44) return;
    float c = 1.0f, sn = 0.0f;
    if (s < SEQ - 1) {
        int fx = s % 24, fy = s / 24;
        int i = (p < 22) ? p : (p - 22);
        float inv = powf(10000.0f, -((2.0f * (float)i) / 44.0f));
        float base = (p < 22) ? (float)(fx + 1) : (float)(fy + 1);
        float fr = base * inv;
        c = cosf(fr); sn = sinf(fr);
    }
    cosT[s * 44 + p] = c;
    sinT[s * 44 + p] = sn;
}

// ---------------------------------------------------------------------------
// QKV GEMM + bias + RoPE, scatter to Q/K [bh][SP][DP] and V^T [bh][DP][SP]
// Block = 8 waves; wave tile 32x64 (2x4 accumulators); 44 K-steps.
// ---------------------------------------------------------------------------
__global__ __launch_bounds__(256) void qkv_gemm_rope_kernel(
        const unsigned short* __restrict__ Ahid,   // [MTOT][HIDDEN] bf16
        const unsigned short* __restrict__ WqkvT,  // [NQKV][HIDDEN] bf16
        const float* __restrict__ bias,            // [NQKV]
        const float* __restrict__ cosT, const float* __restrict__ sinT,
        unsigned short* __restrict__ Qb,
        unsigned short* __restrict__ Kb,
        unsigned short* __restrict__ Vt) {
    const int lane  = threadIdx.x & 31;
    const int wave  = threadIdx.x >> 5;
    const int waveM = wave >> 1;
    const int waveN = wave & 1;
    const int rowBase = blockIdx.x * 128 + waveM * 32;
    const int colBase = blockIdx.y * 128 + waveN * 64;

    v8f zero8 = {0.f,0.f,0.f,0.f,0.f,0.f,0.f,0.f};
    v8f acc[2][4];
#pragma unroll
    for (int i = 0; i < 2; i++)
#pragma unroll
        for (int j = 0; j < 4; j++) acc[i][j] = zero8;

    for (int k = 0; k < HIDDEN; k += 32) {
        v16bf a0 = wmma_ldA(Ahid, HIDDEN, rowBase,      k, lane, MTOT);
        v16bf a1 = wmma_ldA(Ahid, HIDDEN, rowBase + 16, k, lane, MTOT);
#pragma unroll
        for (int j = 0; j < 4; j++) {
            v16bf b = wmma_ldBT(WqkvT, HIDDEN, colBase + j * 16, k, lane);
            acc[0][j] = bf16_wmma(a0, b, acc[0][j]);
            acc[1][j] = bf16_wmma(a1, b, acc[1][j]);
        }
    }

    const int half = lane >> 4, cl = lane & 15;
#pragma unroll
    for (int mt = 0; mt < 2; mt++) {
#pragma unroll
        for (int nt = 0; nt < 4; nt++) {
            const int j = colBase + nt * 16 + cl;
            const float bj = bias[j];
            const int which = j / HIDDEN;          // 0=q 1=k 2=v (uniform per tile)
            const int jj = j - which * HIDDEN;
            const int h = jj / HD;
            const int d = jj - h * HD;
#pragma unroll
            for (int r = 0; r < 8; r++) {
                const int m = rowBase + mt * 16 + half * 8 + r;
                float val = acc[mt][nt][r] + bj;
                // rope pair partner: d parity == lane parity; pair is lanes (2i,2i+1)
                float partner = __shfl_xor(val, 1, 32);
                const bool valid = (m < MTOT);
                const int b = m / SEQ;
                const int s = valid ? (m - b * SEQ) : 0;
                if (which < 2) {
                    const float c  = cosT[s * 44 + (d >> 1)];
                    const float sn = sinT[s * 44 + (d >> 1)];
                    val = val * c + ((d & 1) ? partner * sn : -partner * sn);
                }
                if (valid) {
                    const unsigned short bv = f2bf(val);
                    const size_t bh = (size_t)b * HEADS + h;
                    if (which == 0)      Qb[(bh * SP + s) * DP + d] = bv;
                    else if (which == 1) Kb[(bh * SP + s) * DP + d] = bv;
                    else                 Vt[(bh * DP + d) * SP + s] = bv;
                }
            }
        }
    }
}

// ---------------------------------------------------------------------------
// Flash attention: one wave per (b,h, 16-query tile); 19 x 32-key steps.
// ---------------------------------------------------------------------------
__global__ __launch_bounds__(32) void attn_kernel(
        const unsigned short* __restrict__ Qb,
        const unsigned short* __restrict__ Kb,
        const unsigned short* __restrict__ Vt,
        unsigned short* __restrict__ attnOut) {   // [MTOT][HIDDEN] bf16
    __shared__ unsigned short Plds[16 * 32];
    const int lane = threadIdx.x;
    const int bh = blockIdx.x / QTILES;
    const int qt = blockIdx.x % QTILES;
    const int half = lane >> 4, cl = lane & 15;

    const unsigned short* Qbase = Qb + (size_t)bh * SP * DP + (size_t)qt * 16 * DP;
    const unsigned short* Kbase = Kb + (size_t)bh * SP * DP;
    const unsigned short* Vbase = Vt + (size_t)bh * DP * SP;

    v16bf qa[3];
#pragma unroll
    for (int kd = 0; kd < 3; kd++) qa[kd] = wmma_ldA(Qbase, DP, 0, kd * 32, lane, 1 << 30);

    v8f zero8 = {0.f,0.f,0.f,0.f,0.f,0.f,0.f,0.f};
    v8f O[6];
#pragma unroll
    for (int c = 0; c < 6; c++) O[c] = zero8;
    float mi[8], li[8];
#pragma unroll
    for (int r = 0; r < 8; r++) { mi[r] = -1e30f; li[r] = 0.f; }

    for (int kt = 0; kt < SP; kt += 32) {
        v8f s0 = zero8, s1 = zero8;
#pragma unroll
        for (int kd = 0; kd < 3; kd++) {
            v16bf kb0 = wmma_ldBT(Kbase, DP, kt,      kd * 32, lane);
            v16bf kb1 = wmma_ldBT(Kbase, DP, kt + 16, kd * 32, lane);
            s0 = bf16_wmma(qa[kd], kb0, s0);
            s1 = bf16_wmma(qa[kd], kb1, s1);
        }
        const bool ok0 = (kt + cl)      < SEQ;
        const bool ok1 = (kt + 16 + cl) < SEQ;
#pragma unroll
        for (int r = 0; r < 8; r++) {
            float x0 = ok0 ? s0[r] * SCALE_F : -1e30f;
            float x1 = ok1 ? s1[r] * SCALE_F : -1e30f;
            float mx = fmaxf(x0, x1);
#pragma unroll
            for (int d = 1; d < 16; d <<= 1) mx = fmaxf(mx, __shfl_xor(mx, d, 32));
            const float mnew  = fmaxf(mi[r], mx);
            const float alpha = __expf(mi[r] - mnew);
            const float p0 = __expf(x0 - mnew);
            const float p1 = __expf(x1 - mnew);
            float rs = p0 + p1;
#pragma unroll
            for (int d = 1; d < 16; d <<= 1) rs += __shfl_xor(rs, d, 32);
            li[r] = li[r] * alpha + rs;
            mi[r] = mnew;
#pragma unroll
            for (int c = 0; c < 6; c++) O[c][r] *= alpha;
            Plds[(half * 8 + r) * 32 + cl]      = f2bf(p0);
            Plds[(half * 8 + r) * 32 + 16 + cl] = f2bf(p1);
        }
        __syncthreads();
        v16bf pa;
        {
            const unsigned short* pr = &Plds[cl * 32];
            F16Frag f;
            f.q[0] = *(const uint4*)(pr + 16 * half);
            f.q[1] = *(const uint4*)(pr + 16 * half + 8);
            pa = f.v;
        }
#pragma unroll
        for (int c = 0; c < 6; c++) {
            v16bf vb = wmma_ldBT(Vbase, SP, c * 16, kt, lane);
            O[c] = bf16_wmma(pa, vb, O[c]);
        }
        __syncthreads();
    }

    const int b = bh / HEADS, h = bh % HEADS;
    const int q = qt * 16 + half * 8;
#pragma unroll
    for (int c = 0; c < 6; c++) {
        const int dcol = c * 16 + cl;
#pragma unroll
        for (int r = 0; r < 8; r++) {
            if (q + r < SEQ && dcol < HD) {
                float val = O[c][r] / li[r];
                attnOut[(size_t)(b * SEQ + q + r) * HIDDEN + h * HD + dcol] = f2bf(val);
            }
        }
    }
}

// ---------------------------------------------------------------------------
// Output projection GEMM: attn_bf16 @ woT + b_o -> f32 out
// ---------------------------------------------------------------------------
__global__ __launch_bounds__(256) void oproj_gemm_kernel(
        const unsigned short* __restrict__ Aattn,  // [MTOT][HIDDEN] bf16
        const unsigned short* __restrict__ WoT,    // [HIDDEN][HIDDEN] bf16
        const float* __restrict__ bias,            // [HIDDEN]
        float* __restrict__ out) {                 // [MTOT][HIDDEN] f32
    const int lane  = threadIdx.x & 31;
    const int wave  = threadIdx.x >> 5;
    const int waveM = wave >> 1;
    const int waveN = wave & 1;
    const int rowBase = blockIdx.x * 128 + waveM * 32;
    const int colBase = blockIdx.y * 128 + waveN * 64;

    v8f zero8 = {0.f,0.f,0.f,0.f,0.f,0.f,0.f,0.f};
    v8f acc[2][4];
#pragma unroll
    for (int i = 0; i < 2; i++)
#pragma unroll
        for (int j = 0; j < 4; j++) acc[i][j] = zero8;

    for (int k = 0; k < HIDDEN; k += 32) {
        v16bf a0 = wmma_ldA(Aattn, HIDDEN, rowBase,      k, lane, MTOT);
        v16bf a1 = wmma_ldA(Aattn, HIDDEN, rowBase + 16, k, lane, MTOT);
#pragma unroll
        for (int j = 0; j < 4; j++) {
            v16bf b = wmma_ldBT(WoT, HIDDEN, colBase + j * 16, k, lane);
            acc[0][j] = bf16_wmma(a0, b, acc[0][j]);
            acc[1][j] = bf16_wmma(a1, b, acc[1][j]);
        }
    }

    const int half = lane >> 4, cl = lane & 15;
#pragma unroll
    for (int mt = 0; mt < 2; mt++) {
#pragma unroll
        for (int nt = 0; nt < 4; nt++) {
            const int j = colBase + nt * 16 + cl;
            const float bj = bias[j];
#pragma unroll
            for (int r = 0; r < 8; r++) {
                const int m = rowBase + mt * 16 + half * 8 + r;
                if (m < MTOT)
                    out[(size_t)m * HIDDEN + j] = acc[mt][nt][r] + bj;
            }
        }
    }
}

// ---------------------------------------------------------------------------
extern "C" void kernel_launch(void* const* d_in, const int* in_sizes, int n_in,
                              void* d_out, int out_size, void* d_ws, size_t ws_size,
                              hipStream_t stream) {
    (void)in_sizes; (void)n_in; (void)out_size; (void)ws_size;
    const float* hs   = (const float*)d_in[0];
    const float* wqkv = (const float*)d_in[1];
    const float* bqkv = (const float*)d_in[2];
    const float* wo   = (const float*)d_in[3];
    const float* bo   = (const float*)d_in[4];
    float* out = (float*)d_out;

    unsigned short* ws = (unsigned short*)d_ws;
    size_t off = 0;
    unsigned short* hid  = ws + off; off += (size_t)MTOT * HIDDEN;
    unsigned short* wqT  = ws + off; off += (size_t)NQKV * HIDDEN;
    unsigned short* woT  = ws + off; off += (size_t)HIDDEN * HIDDEN;
    unsigned short* Qb   = ws + off; off += (size_t)BATCH * HEADS * SP * DP;
    unsigned short* Kb   = ws + off; off += (size_t)BATCH * HEADS * SP * DP;
    unsigned short* Vt   = ws + off; off += (size_t)BATCH * HEADS * SP * DP;
    unsigned short* attn = ws + off; off += (size_t)MTOT * HIDDEN;
    float* cosT = (float*)(ws + off); off += (size_t)SP * 44 * 2;
    float* sinT = (float*)(ws + off); off += (size_t)SP * 44 * 2;

    // 1) zero V (its padded region must not inject NaNs through P==0 * pad)
    size_t vN16 = ((size_t)BATCH * HEADS * SP * DP) / 8;
    zero_kernel<<<4096, 256, 0, stream>>>((uint4*)Vt, vN16);

    // 2) hidden f32 -> bf16
    int n4 = (MTOT * HIDDEN) / 4;
    cvt_hidden_kernel<<<(n4 + 255) / 256, 256, 0, stream>>>(hs, hid, n4);

    // 3) weights: transpose + convert to bf16 [N][K]
    transpose_cvt_kernel<<<dim3(NQKV / 32, HIDDEN / 32), dim3(32, 8), 0, stream>>>(
        wqkv, wqT, HIDDEN, NQKV);
    transpose_cvt_kernel<<<dim3(HIDDEN / 32, HIDDEN / 32), dim3(32, 8), 0, stream>>>(
        wo, woT, HIDDEN, HIDDEN);

    // 4) rope tables
    rope_table_kernel<<<SP, 64, 0, stream>>>(cosT, sinT);

    // 5) QKV projection + rope
    qkv_gemm_rope_kernel<<<dim3((MTOT + 127) / 128, NQKV / 128), 256, 0, stream>>>(
        hid, wqT, bqkv, cosT, sinT, Qb, Kb, Vt);

    // 6) flash attention
    attn_kernel<<<BATCH * HEADS * QTILES, 32, 0, stream>>>(Qb, Kb, Vt, attn);

    // 7) output projection
    oproj_gemm_kernel<<<dim3((MTOT + 127) / 128, HIDDEN / 128), 256, 0, stream>>>(
        attn, woT, bo, out);
}
